// NisuyNN_32615981646044
// MI455X (gfx1250) — matrix operand order repaired and checked by reading, not compiled, not run on hardware
//
#include <hip/hip_runtime.h>
#include <hip/hip_bf16.h>

// ---------------------------------------------------------------------------
// Types for gfx1250 WMMA
// ---------------------------------------------------------------------------
typedef float  v8f_t   __attribute__((ext_vector_type(8)));
typedef __bf16 v16bf_t __attribute__((ext_vector_type(16)));

static __device__ __forceinline__ unsigned short f32_to_bf16_rne(float f) {
  union { float f; unsigned int u; } v; v.f = f;
  unsigned int u = v.u;
  u += 0x7fffu + ((u >> 16) & 1u);   // round-to-nearest-even
  return (unsigned short)(u >> 16);
}

struct Bits256 { uint4 lo, hi; };
static __device__ __forceinline__ v16bf_t load_frag(const unsigned short* p0,
                                                    const unsigned short* p1) {
  Bits256 t;
  t.lo = *reinterpret_cast<const uint4*>(p0);
  t.hi = *reinterpret_cast<const uint4*>(p1);
  return __builtin_bit_cast(v16bf_t, t);
}

// ---------------------------------------------------------------------------
// Transpose + convert: src fp32 (R x C, ld=ldsrc) -> dst bf16 (C x R, ld=lddst)
// Weights stored N-major so B-fragments are contiguous-K 16B loads.
// ---------------------------------------------------------------------------
__global__ __launch_bounds__(256) void transpose_to_bf16(
    const float* __restrict__ src, int R, int C, long ldsrc,
    unsigned short* __restrict__ dst, long lddst) {
  __shared__ float tile[32][33];
  const int tx = threadIdx.x, ty = threadIdx.y;
  const int cx = blockIdx.x * 32 + tx;
#pragma unroll
  for (int k = 0; k < 4; ++k) {
    const int ry = blockIdx.y * 32 + ty + k * 8;
    if (ry < R && cx < C) tile[ty + k * 8][tx] = src[(long)ry * ldsrc + cx];
  }
  __syncthreads();
  const int rr = blockIdx.y * 32 + tx;   // dst column (= src row)
#pragma unroll
  for (int k = 0; k < 4; ++k) {
    const int cc = blockIdx.x * 32 + ty + k * 8;  // dst row (= src col)
    if (cc < C && rr < R)
      dst[(long)cc * lddst + rr] = f32_to_bf16_rne(tile[tx][ty + k * 8]);
  }
}

// fp32 -> bf16 elementwise (node embeddings as GEMM A matrix)
__global__ void convert_to_bf16(const float* __restrict__ src,
                                unsigned short* __restrict__ dst, int n) {
  const int i = blockIdx.x * blockDim.x + threadIdx.x;
  if (i < n) dst[i] = f32_to_bf16_rne(src[i]);
}

// ---------------------------------------------------------------------------
// bf16 WMMA GEMM:  C(MxN) = A(MxK,bf16,row-major) * Bt(NxK,bf16,N-major)
//   block = 256 threads = 8 waves, wave tile = 64x32 (4x2 WMMA tiles),
//   block tile = 128x128.  M,N multiples of 128; K multiple of 32.
// mode 0: store fp32 raw;  mode 1: bias+relu -> bf16;
// mode 2: bias+sigmoid, then *mult[row&7]+add[row&7] -> fp32
// ---------------------------------------------------------------------------
__global__ __launch_bounds__(256) void gemm_bf16_wmma(
    const unsigned short* __restrict__ A, long lda,
    const unsigned short* __restrict__ Bt, long ldb,
    void* __restrict__ Cout, long ldc, int K,
    const float* __restrict__ bias, int mode,
    const float* __restrict__ mult, const float* __restrict__ addc) {
  const int lane = (int)(threadIdx.x & 31u);
  const int wid  = (int)(threadIdx.x >> 5u);
  const int lr   = lane & 15;
  const int lh   = lane >> 4;
  const long m_base = (long)blockIdx.y * 128 + (long)(wid >> 2) * 64;
  const long n_base = (long)blockIdx.x * 128 + (long)(wid & 3) * 32;

  v8f_t acc[4][2] = {};

  for (int k0 = 0; k0 < K; k0 += 32) {
    // A fragments (16x32 bf16): lanes 0-15 -> K 0-7 & 16-23, lanes 16-31 -> +8
    v16bf_t af[4];
#pragma unroll
    for (int i = 0; i < 4; ++i) {
      const unsigned short* pa = A + (m_base + i * 16 + lr) * lda + (k0 + lh * 8);
      af[i] = load_frag(pa, pa + 16);
    }
    // B fragments (32x16 bf16) from N-major weights: lanes 0-15 -> K 0-7 & 16-23, lanes 16-31 -> +8
    v16bf_t bfr[2];
#pragma unroll
    for (int j = 0; j < 2; ++j) {
      const unsigned short* pb = Bt + (n_base + j * 16 + lr) * ldb + (k0 + lh * 8);
      bfr[j] = load_frag(pb, pb + 16);
    }
#pragma unroll
    for (int i = 0; i < 4; ++i)
#pragma unroll
      for (int j = 0; j < 2; ++j)
        acc[i][j] = __builtin_amdgcn_wmma_f32_16x16x32_bf16(
            false, af[i], false, bfr[j], (short)0, acc[i][j], false, false);
  }

  // Epilogue: C/D layout -> row = m0 + lh*8 + v, col = n0 + lr
#pragma unroll
  for (int i = 0; i < 4; ++i) {
#pragma unroll
    for (int j = 0; j < 2; ++j) {
      const long col = n_base + j * 16 + lr;
      const float bcol = (mode != 0) ? bias[col] : 0.0f;
#pragma unroll
      for (int v = 0; v < 8; ++v) {
        const long row = m_base + i * 16 + lh * 8 + v;
        float x = acc[i][j][v];
        if (mode == 0) {
          reinterpret_cast<float*>(Cout)[row * ldc + col] = x;
        } else if (mode == 1) {
          x += bcol;
          x = x > 0.0f ? x : 0.0f;
          reinterpret_cast<unsigned short*>(Cout)[row * ldc + col] = f32_to_bf16_rne(x);
        } else {
          x += bcol;
          x = 1.0f / (1.0f + __expf(-x));
          const int bb = (int)(row & 7);
          x = x * mult[bb] + addc[bb];
          reinterpret_cast<float*>(Cout)[row * ldc + col] = x;
        }
      }
    }
  }
}

// ---------------------------------------------------------------------------
// G[b, col] = sum_k node_flat[b,k] * W1[256+k, col]   (8 x 4096 x 4096, fp32)
// ---------------------------------------------------------------------------
__global__ __launch_bounds__(256) void g_kernel(const float* __restrict__ emb,
                                                const float* __restrict__ W1,
                                                float* __restrict__ G) {
  const int col = blockIdx.x * blockDim.x + threadIdx.x;  // 0..4095
  float acc[8] = {0, 0, 0, 0, 0, 0, 0, 0};
  for (int k = 0; k < 4096; ++k) {
    const float w = W1[(long)(256 + k) * 4096 + col];
#pragma unroll
    for (int b = 0; b < 8; ++b) acc[b] += emb[b * 4096 + k] * w;
  }
#pragma unroll
  for (int b = 0; b < 8; ++b) G[(long)b * 4096 + col] = acc[b];
}

// ---------------------------------------------------------------------------
// h1[r,c] = relu(S[b*32+s,c] + V[b*32+t,c] + G[b,c] + b1[c]) -> bf16
//   r = p*8 + b, p = s*32 + t
// ---------------------------------------------------------------------------
__global__ __launch_bounds__(256) void assemble_h1(
    const float* __restrict__ S, const float* __restrict__ V,
    const float* __restrict__ G, const float* __restrict__ b1,
    unsigned short* __restrict__ h1) {
  const int r = blockIdx.x;
  const int p = r >> 3, b = r & 7;
  const int s = p >> 5, t = p & 31;
  const float* Sr = S + (long)(b * 32 + s) * 4096;
  const float* Vr = V + (long)(b * 32 + t) * 4096;
  const float* Gr = G + (long)b * 4096;
  unsigned short* out = h1 + (long)r * 4096;
  for (int c = threadIdx.x; c < 4096; c += 256) {
    float x = Sr[c] + Vr[c] + Gr[c] + b1[c];
    out[c] = f32_to_bf16_rne(x > 0.0f ? x : 0.0f);
  }
}

// ---------------------------------------------------------------------------
// 50-step power iteration on 32x32 matrices, one wave32 per matrix.
// lane i holds row i in registers; b broadcast via shuffles.
// delta[r*32 + i] = b_i * Ts[b, s, t] / b_s
// ---------------------------------------------------------------------------
__global__ __launch_bounds__(256) void power_iter(
    const float* __restrict__ policy, const float* __restrict__ Ts,
    float* __restrict__ delta) {
  const int wid  = (int)(threadIdx.x >> 5u);
  const int lane = (int)(threadIdx.x & 31u);
  const int r = blockIdx.x * 8 + wid;  // matrix index (p*8 + b)
  const float* A = policy + (long)r * 1024;
  float a[32];
#pragma unroll
  for (int j4 = 0; j4 < 8; ++j4) {
    const float4 v = *reinterpret_cast<const float4*>(A + lane * 32 + j4 * 4);
    a[j4 * 4 + 0] = v.x; a[j4 * 4 + 1] = v.y;
    a[j4 * 4 + 2] = v.z; a[j4 * 4 + 3] = v.w;
  }
  float bvec = 1.0f / 32.0f;
  for (int it = 0; it < 50; ++it) {
    float nb = 0.0f;
#pragma unroll
    for (int j = 0; j < 32; ++j) nb += a[j] * __shfl(bvec, j, 32);
    float ss = nb * nb;
#pragma unroll
    for (int off = 16; off > 0; off >>= 1) ss += __shfl_xor(ss, off, 32);
    bvec = nb * rsqrtf(ss);
  }
  const int p = r >> 3, bidx = r & 7;
  const int s = p >> 5, t = p & 31;
  const float src = __shfl(bvec, s, 32);
  const float tst = Ts[bidx * 1024 + s * 32 + t];
  delta[(long)r * 32 + lane] = bvec * (tst / src);
}

// out[b*32 + i] = sum_p delta[p*256 + b*32 + i]   (deterministic, no atomics)
__global__ __launch_bounds__(256) void reduce_delta(
    const float* __restrict__ delta, float* __restrict__ out) {
  const int tid = threadIdx.x;  // 0..255
  float s = 0.0f;
  for (int p = 0; p < 1024; ++p) s += delta[(long)p * 256 + tid];
  out[tid] = s;
}

// ---------------------------------------------------------------------------
// Workspace layout (bytes). policy reuses h1; delta reuses h2.  ~145 MB total,
// fits the 192 MB L2 working-set target.
// ---------------------------------------------------------------------------
static const size_t OFF_WT1A = 0;                                        // 4096x128 bf16
static const size_t OFF_WT1B = OFF_WT1A + (size_t)4096 * 128 * 2;
static const size_t OFF_WT2  = OFF_WT1B + (size_t)4096 * 128 * 2;        // 2048x4096 bf16
static const size_t OFF_WT3  = OFF_WT2  + (size_t)2048 * 4096 * 2;       // 1024x2048 bf16
static const size_t OFF_WT4  = OFF_WT3  + (size_t)1024 * 2048 * 2;       // 1024x1024 bf16
static const size_t OFF_EBF  = OFF_WT4  + (size_t)1024 * 1024 * 2;       // 256x128 bf16
static const size_t OFF_S    = OFF_EBF  + (size_t)256 * 128 * 2;         // 256x4096 f32
static const size_t OFF_V    = OFF_S    + (size_t)256 * 4096 * 4;        // 256x4096 f32
static const size_t OFF_G    = OFF_V    + (size_t)256 * 4096 * 4;        // 8x4096 f32
static const size_t OFF_H1   = OFF_G    + (size_t)8 * 4096 * 4;          // 8192x4096 bf16 (reused: policy f32)
static const size_t OFF_H2   = OFF_H1   + (size_t)8192 * 4096 * 2;       // 8192x2048 bf16 (reused: delta f32)
static const size_t OFF_H3   = OFF_H2   + (size_t)8192 * 2048 * 2;       // 8192x1024 bf16

extern "C" void kernel_launch(void* const* d_in, const int* in_sizes, int n_in,
                              void* d_out, int out_size, void* d_ws, size_t ws_size,
                              hipStream_t stream) {
  (void)in_sizes; (void)n_in; (void)out_size; (void)ws_size;
  const float* emb  = (const float*)d_in[0];   // (8,32,128)
  const float* Ts   = (const float*)d_in[2];   // (8,32,32)
  const float* mult = (const float*)d_in[3];   // (8,1,1)
  const float* addc = (const float*)d_in[4];   // (8,1,1)
  const float* W1 = (const float*)d_in[5];     // (4352,4096)
  const float* b1 = (const float*)d_in[6];
  const float* W2 = (const float*)d_in[7];     // (4096,2048)
  const float* b2 = (const float*)d_in[8];
  const float* W3 = (const float*)d_in[9];     // (2048,1024)
  const float* b3 = (const float*)d_in[10];
  const float* W4 = (const float*)d_in[11];    // (1024,1024)
  const float* b4 = (const float*)d_in[12];

  char* ws = (char*)d_ws;
  unsigned short* Wt1a = (unsigned short*)(ws + OFF_WT1A);
  unsigned short* Wt1b = (unsigned short*)(ws + OFF_WT1B);
  unsigned short* Wt2  = (unsigned short*)(ws + OFF_WT2);
  unsigned short* Wt3  = (unsigned short*)(ws + OFF_WT3);
  unsigned short* Wt4  = (unsigned short*)(ws + OFF_WT4);
  unsigned short* Ebf  = (unsigned short*)(ws + OFF_EBF);
  float* S      = (float*)(ws + OFF_S);
  float* V      = (float*)(ws + OFF_V);
  float* G      = (float*)(ws + OFF_G);
  unsigned short* h1 = (unsigned short*)(ws + OFF_H1);
  unsigned short* h2 = (unsigned short*)(ws + OFF_H2);
  unsigned short* h3 = (unsigned short*)(ws + OFF_H3);
  float* policy = (float*)(ws + OFF_H1);   // reuse (h1 dead after layer 2)
  float* delta  = (float*)(ws + OFF_H2);   // reuse (h2 dead after layer 3)

  const dim3 tb(32, 8);
  // Weight transpose+convert to N-major bf16
  transpose_to_bf16<<<dim3(128, 4),  tb, 0, stream>>>(W1,              128, 4096, 4096, Wt1a, 128);
  transpose_to_bf16<<<dim3(128, 4),  tb, 0, stream>>>(W1 + 128 * 4096, 128, 4096, 4096, Wt1b, 128);
  transpose_to_bf16<<<dim3(64, 128), tb, 0, stream>>>(W2, 4096, 2048, 2048, Wt2, 4096);
  transpose_to_bf16<<<dim3(32, 64),  tb, 0, stream>>>(W3, 2048, 1024, 1024, Wt3, 2048);
  transpose_to_bf16<<<dim3(32, 32),  tb, 0, stream>>>(W4, 1024, 1024, 1024, Wt4, 1024);
  convert_to_bf16<<<128, 256, 0, stream>>>(emb, Ebf, 256 * 128);

  // Structured layer 1: node-broadcast term (fp32 VALU, exact)
  g_kernel<<<16, 256, 0, stream>>>(emb, W1, G);
  // s/t embedding terms: (256 x 128) @ (128 x 4096), WMMA, fp32 out
  gemm_bf16_wmma<<<dim3(32, 2), 256, 0, stream>>>(Ebf, 128, Wt1a, 128, S, 4096, 128,
                                                  nullptr, 0, nullptr, nullptr);
  gemm_bf16_wmma<<<dim3(32, 2), 256, 0, stream>>>(Ebf, 128, Wt1b, 128, V, 4096, 128,
                                                  nullptr, 0, nullptr, nullptr);
  // h1 = relu(S + V + G + b1) -> bf16
  assemble_h1<<<8192, 256, 0, stream>>>(S, V, G, b1, h1);

  // Layers 2..4 (WMMA, fused epilogues)
  gemm_bf16_wmma<<<dim3(16, 64), 256, 0, stream>>>(h1, 4096, Wt2, 4096, h2, 2048, 4096,
                                                   b2, 1, nullptr, nullptr);
  gemm_bf16_wmma<<<dim3(8, 64), 256, 0, stream>>>(h2, 2048, Wt3, 2048, h3, 1024, 2048,
                                                  b3, 1, nullptr, nullptr);
  gemm_bf16_wmma<<<dim3(8, 64), 256, 0, stream>>>(h3, 1024, Wt4, 1024, policy, 1024, 1024,
                                                  b4, 2, mult, addc);

  // Power iteration (one wave per 32x32 matrix) + deterministic reduction
  power_iter<<<1024, 256, 0, stream>>>(policy, Ts, delta);
  reduce_delta<<<1, 256, 0, stream>>>(delta, (float*)d_out);
}
